// HymbaBlock_13297218748942
// MI455X (gfx1250) — matrix operand order repaired and checked by reading, not compiled, not run on hardware
//
#include <hip/hip_runtime.h>
#include <hip/hip_bf16.h>
#include <math.h>
#include <stdint.h>

// ---------------- model dims ----------------
#define BATCH   2
#define S_LEN   2048
#define TOK     4096          // BATCH*S_LEN
#define HID_    1024
#define INTER_  2048
#define LT_     5888          // LATENT + INTER
#define XS_OFF  1792
#define GATE_OFF 3840
#define NST     16
#define NH      16
#define HD      64
#define VHD     128

typedef __attribute__((ext_vector_type(16))) __bf16 v16bf;
typedef __attribute__((ext_vector_type(8)))  float  v8f;

union Frag { v16bf v; uint4 q[2]; };
union U4H8 { uint4 q; __bf16 h[8]; };

__device__ inline v8f v8f_zero() {
  v8f z;
  for (int i = 0; i < 8; ++i) z[i] = 0.f;
  return z;
}

__device__ inline v8f wmma_bf16(const Frag& a, const Frag& b, v8f c) {
  return __builtin_amdgcn_wmma_f32_16x16x32_bf16(
      false, a.v, false, b.v, (short)0, c, false, false);
}

// CDNA5 async copy: 16B global -> LDS, tracked by ASYNCcnt (no VGPR data path).
// ldsOff = LDS byte address (low 32 bits of flat shared pointer), g = 64-bit global addr.
__device__ inline void async_ld16(unsigned ldsOff, const void* g) {
  asm volatile("global_load_async_to_lds_b128 %0, %1, off"
               :: "v"(ldsOff), "v"(g) : "memory");
}
__device__ inline void wait_async0() {
  asm volatile("s_wait_asynccnt 0" ::: "memory");
}
__device__ inline unsigned lds_addr(const void* p) {
  return (unsigned)(uintptr_t)p;
}

// ================= generic bf16 WMMA GEMM: C[M,N] = A[M,K] @ B[N,K]^T =================
// M%128==0, N%128==0, K%32==0.  256 threads = 8 waves, wave tile 64x32 (4x2 WMMA tiles).
// Tiles staged with GLOBAL_LOAD_ASYNC_TO_LDS_B128 (double-buffered, ASYNCcnt-synced).
__global__ __launch_bounds__(256)
void gemm_bf16_tn(const __bf16* __restrict__ A, const __bf16* __restrict__ B,
                  float* __restrict__ C, int M, int N, int K)
{
  (void)M;
  __shared__ __bf16 sA[2][128 * 40];   // 32 cols + 8 pad (80B rows, 16B aligned)
  __shared__ __bf16 sB[2][128 * 40];
  const int t = threadIdx.x;
  const int lane = t & 31, w = t >> 5;
  const int wm = w >> 2, wn = w & 3;
  const int hb = lane >> 4, l15 = lane & 15;
  const int mBase = blockIdx.y * 128;
  const int nBase = blockIdx.x * 128;
  const int nk = K >> 5;

  // loader: 512 uint4 per matrix tile, 2 per thread
  const int u0 = t, u1 = t + 256;
  const int ar0 = u0 >> 2, ac0 = (u0 & 3) * 8;
  const int ar1 = u1 >> 2, ac1 = (u1 & 3) * 8;

  // per-thread LDS byte addresses (buffer 0)
  const unsigned BUFB = 128 * 40 * 2;  // bytes per buffer
  const unsigned la0 = lds_addr(&sA[0][ar0 * 40 + ac0]);
  const unsigned la1 = lds_addr(&sA[0][ar1 * 40 + ac1]);
  const unsigned lb0 = lds_addr(&sB[0][ar0 * 40 + ac0]);
  const unsigned lb1 = lds_addr(&sB[0][ar1 * 40 + ac1]);
  // per-thread global source pointers (k-tile 0)
  const __bf16* gA0 = A + (size_t)(mBase + ar0) * K + ac0;
  const __bf16* gA1 = A + (size_t)(mBase + ar1) * K + ac1;
  const __bf16* gB0 = B + (size_t)(nBase + ar0) * K + ac0;
  const __bf16* gB1 = B + (size_t)(nBase + ar1) * K + ac1;

  v8f acc[4][2];
#pragma unroll
  for (int i = 0; i < 4; ++i)
#pragma unroll
    for (int j = 0; j < 2; ++j) acc[i][j] = v8f_zero();

  // prime buffer 0
  async_ld16(la0, gA0);
  async_ld16(la1, gA1);
  async_ld16(lb0, gB0);
  async_ld16(lb1, gB1);
  wait_async0();
  __syncthreads();

  for (int kt = 0; kt < nk; ++kt) {
    const int cur = kt & 1;
    if (kt + 1 < nk) {                 // kick async fill of the other buffer
      const unsigned o = (unsigned)(cur ^ 1) * BUFB;
      const int k1 = (kt + 1) << 5;
      async_ld16(la0 + o, gA0 + k1);
      async_ld16(la1 + o, gA1 + k1);
      async_ld16(lb0 + o, gB0 + k1);
      async_ld16(lb1 + o, gB1 + k1);
      if (kt + 2 < nk) {               // global_prefetch_b8, two tiles ahead
        const int k2 = (kt + 2) << 5;
        __builtin_prefetch(gA0 + k2, 0, 1);
        __builtin_prefetch(gB0 + k2, 0, 1);
      }
    }
    Frag fa[4], fb[2];
#pragma unroll
    for (int i = 0; i < 4; ++i) {      // A frag: row=l15, K: hb*8..+7 and 16+hb*8..+7
      const __bf16* p = &sA[cur][(wm * 64 + i * 16 + l15) * 40 + hb * 8];
      fa[i].q[0] = *(const uint4*)p;
      fa[i].q[1] = *(const uint4*)(p + 16);
    }
#pragma unroll
    for (int j = 0; j < 2; ++j) {      // B frag: col=l15, K: hb*8..+7 and 16+hb*8..+7
      const __bf16* p = &sB[cur][(wn * 32 + j * 16 + l15) * 40 + hb * 8];
      fb[j].q[0] = *(const uint4*)p;
      fb[j].q[1] = *(const uint4*)(p + 16);
    }
#pragma unroll
    for (int i = 0; i < 4; ++i)
#pragma unroll
      for (int j = 0; j < 2; ++j)
        acc[i][j] = wmma_bf16(fa[i], fb[j], acc[i][j]);

    if (kt + 1 < nk) {
      wait_async0();                   // this wave's async fills done
      __syncthreads();                 // all waves' fills done
    }
  }
#pragma unroll
  for (int i = 0; i < 4; ++i)
#pragma unroll
    for (int j = 0; j < 2; ++j)
#pragma unroll
      for (int r = 0; r < 8; ++r) {
        int row = mBase + wm * 64 + i * 16 + r + 8 * hb;
        int col = nBase + wn * 32 + j * 16 + l15;
        C[(size_t)row * N + col] = acc[i][j][r];
      }
}

// ================= packs / converts =================
__global__ void cvt_bf16(const float* __restrict__ s, __bf16* __restrict__ d, size_t n) {
  size_t i = (size_t)blockIdx.x * 256 + threadIdx.x;
  if (i < n) d[i] = (__bf16)s[i];
}
__global__ void pack_wx(const float* __restrict__ s, __bf16* __restrict__ d) {
  int i = blockIdx.x * 256 + threadIdx.x;        // 128*2048
  int r = i >> 11;
  d[i] = (r < 96) ? (__bf16)s[i] : (__bf16)0.f;
}
__global__ void pack_qkv(const float* __restrict__ proj, __bf16* __restrict__ qkv) {
  size_t i = (size_t)blockIdx.x * 256 + threadIdx.x;   // TOK*1792
  size_t tok = i / 1792; int col = (int)(i - tok * 1792);
  float v = proj[tok * LT_ + col];
  if (col < 1024) v *= 0.125f;                   // fold 1/sqrt(64) into q
  qkv[i] = (__bf16)v;
}

// ================= depthwise causal conv + SiLU =================
__global__ void conv_silu(const float* __restrict__ proj, const float* __restrict__ cw,
                          const float* __restrict__ cb, float* __restrict__ xc) {
  int d = blockIdx.x * 256 + threadIdx.x;
  int s = blockIdx.y, b = blockIdx.z;
  float a = cb[d];
#pragma unroll
  for (int j = 0; j < 4; ++j) {
    int ss = s - 3 + j;
    if (ss >= 0) a += proj[(size_t)(b * S_LEN + ss) * LT_ + XS_OFF + d] * cw[d * 4 + j];
  }
  xc[(size_t)(b * S_LEN + s) * INTER_ + d] = a / (1.f + __expf(-a));
}

// ================= segment rmsnorms on ssm_p =================
__global__ void dtbc_norm(const float* __restrict__ ssmp,
                          const float* __restrict__ wdt, const float* __restrict__ wb,
                          const float* __restrict__ wc,
                          float* __restrict__ dtn, float* __restrict__ bn,
                          float* __restrict__ cn) {
  __shared__ float red[96];
  __shared__ float scale[3];
  int tok = blockIdx.x, t = threadIdx.x;
  const float* p = ssmp + (size_t)tok * 128;
  float v = (t < 96) ? p[t] : 0.f;
  if (t < 96) red[t] = v * v;
  __syncthreads();
  if (t == 0) { float s = 0; for (int i = 0;  i < 64; ++i) s += red[i]; scale[0] = rsqrtf(s / 64.f + 1e-6f); }
  if (t == 1) { float s = 0; for (int i = 64; i < 80; ++i) s += red[i]; scale[1] = rsqrtf(s / 16.f + 1e-6f); }
  if (t == 2) { float s = 0; for (int i = 80; i < 96; ++i) s += red[i]; scale[2] = rsqrtf(s / 16.f + 1e-6f); }
  __syncthreads();
  if (t < 64)       dtn[(size_t)tok * 64 + t]        = v * scale[0] * wdt[t];
  else if (t < 80)  bn [(size_t)tok * 16 + (t - 64)] = v * scale[1] * wb[t - 64];
  else if (t < 96)  cn [(size_t)tok * 16 + (t - 80)] = v * scale[2] * wc[t - 80];
}

// ================= delta = softplus(dt @ W_dt^T + b_dt) =================
__global__ void delta_kernel(const float* __restrict__ dtn, const float* __restrict__ wdt,
                             const float* __restrict__ bdt, float* __restrict__ delta) {
  __shared__ float sdt[64];
  int tok = blockIdx.x, t = threadIdx.x;
  if (t < 64) sdt[t] = dtn[(size_t)tok * 64 + t];
  __syncthreads();
#pragma unroll
  for (int ii = 0; ii < 8; ++ii) {
    int i = t + ii * 256;
    float a = bdt[i];
    const float* wr = wdt + (size_t)i * 64;
#pragma unroll 8
    for (int k = 0; k < 64; ++k) a += sdt[k] * wr[k];
    delta[(size_t)tok * INTER_ + i] = (a > 20.f) ? a : log1pf(__expf(a));
  }
}

// ================= selective scan (sequential in S) + gate =================
__global__ void scan_kernel(const float* __restrict__ delta, const float* __restrict__ bn,
                            const float* __restrict__ cn, const float* __restrict__ xc,
                            const float* __restrict__ proj, const float* __restrict__ alog,
                            const float* __restrict__ dp, float* __restrict__ y) {
  __shared__ float sB[16], sC[16];
  int d = blockIdx.x * 256 + threadIdx.x;
  int b = blockIdx.y, t = threadIdx.x;
  float A[NST], h[NST];
#pragma unroll
  for (int n = 0; n < NST; ++n) { A[n] = -__expf(alog[(size_t)d * NST + n]); h[n] = 0.f; }
  float D = dp[d];
  for (int s = 0; s < S_LEN; ++s) {
    size_t tok = (size_t)b * S_LEN + s;
    if (t < 16)      sB[t]      = bn[tok * 16 + t];
    else if (t < 32) sC[t - 16] = cn[tok * 16 + (t - 16)];
    __syncthreads();
    float dt = delta[tok * INTER_ + d];
    float xv = xc[tok * INTER_ + d];
    float g  = proj[tok * LT_ + GATE_OFF + d];
    float dx = dt * xv, yv = 0.f;
#pragma unroll
    for (int n = 0; n < NST; ++n) {
      h[n] = h[n] * __expf(dt * A[n]) + dx * sB[n];
      yv += h[n] * sC[n];
    }
    y[tok * INTER_ + d] = (yv + xv * D) * (g / (1.f + __expf(-g)));
    __syncthreads();
  }
}

// ================= flash attention (bf16 WMMA, fp32 softmax) =================
__global__ __launch_bounds__(128)
void attn_kernel(const __bf16* __restrict__ qkv, float* __restrict__ out) {
  __shared__ __bf16 sK[64 * 72];       // [key][hdim], pad to 144B rows
  __shared__ __bf16 sVt[128 * 72];     // [vdim][key]
  __shared__ __bf16 sP[4][16 * 72];    // per-wave P staging
  const int t = threadIdx.x;
  const int lane = t & 31, w = t >> 5;
  const int hb = lane >> 4, l15 = lane & 15;
  const int qb = blockIdx.x, head = blockIdx.y, b = blockIdx.z;
  const int kvh = head >> 2;

  Frag fq[2];
  {
    const size_t tok = (size_t)b * S_LEN + qb * 64 + w * 16 + l15;
    const __bf16* qp = qkv + tok * 1792 + head * 64;
#pragma unroll
    for (int kc = 0; kc < 2; ++kc) {
      fq[kc].q[0] = *(const uint4*)(qp + kc * 32 + hb * 8);
      fq[kc].q[1] = *(const uint4*)(qp + kc * 32 + 16 + hb * 8);
    }
  }
  v8f acc[8];
#pragma unroll
  for (int v = 0; v < 8; ++v) acc[v] = v8f_zero();
  float mrow[8], lrow[8];
#pragma unroll
  for (int r = 0; r < 8; ++r) { mrow[r] = -__builtin_inff(); lrow[r] = 0.f; }

  for (int kb = 0; kb <= qb; ++kb) {
#pragma unroll
    for (int ii = 0; ii < 4; ++ii) {                       // K tile 64x64 via async-to-LDS
      int u = t * 4 + ii, ky = u >> 3, cu = (u & 7) * 8;
      async_ld16(lds_addr(&sK[ky * 72 + cu]),
                 qkv + ((size_t)b * S_LEN + kb * 64 + ky) * 1792 + 1024 + kvh * 64 + cu);
    }
#pragma unroll
    for (int ii = 0; ii < 8; ++ii) {                       // V tile 64x128, transposed
      int u = t * 8 + ii, ky = u >> 4, cu = (u & 15) * 8;
      U4H8 vv;
      vv.q = *(const uint4*)(qkv + ((size_t)b * S_LEN + kb * 64 + ky) * 1792 + 1280 + kvh * 128 + cu);
#pragma unroll
      for (int jj = 0; jj < 8; ++jj) sVt[(cu + jj) * 72 + ky] = vv.h[jj];
    }
    wait_async0();
    __syncthreads();

    v8f st[4];
#pragma unroll
    for (int nt = 0; nt < 4; ++nt) {                       // scores 16x64
      Frag fk0, fk1;
      const __bf16* p0 = &sK[(nt * 16 + l15) * 72 + hb * 8];
      fk0.q[0] = *(const uint4*)p0; fk0.q[1] = *(const uint4*)(p0 + 16);
      const __bf16* p1 = &sK[(nt * 16 + l15) * 72 + 32 + hb * 8];
      fk1.q[0] = *(const uint4*)p1; fk1.q[1] = *(const uint4*)(p1 + 16);
      v8f sc = v8f_zero();
      sc = wmma_bf16(fq[0], fk0, sc);
      sc = wmma_bf16(fq[1], fk1, sc);
      st[nt] = sc;
    }
    if (kb == qb) {                                        // causal mask on diagonal block
#pragma unroll
      for (int nt = 0; nt < 4; ++nt)
#pragma unroll
        for (int r = 0; r < 8; ++r) {
          int rowg = qb * 64 + w * 16 + r + 8 * hb;
          int colg = kb * 64 + nt * 16 + l15;
          if (colg > rowg) st[nt][r] = -1e30f;
        }
    }
#pragma unroll
    for (int r = 0; r < 8; ++r) {                          // online softmax per row
      float mx = fmaxf(fmaxf(st[0][r], st[1][r]), fmaxf(st[2][r], st[3][r]));
#pragma unroll
      for (int off = 1; off < 16; off <<= 1) mx = fmaxf(mx, __shfl_xor(mx, off, 32));
      float nm = fmaxf(mrow[r], mx);
      float sc = __expf(mrow[r] - nm);
      mrow[r] = nm;
      float rs = 0.f;
#pragma unroll
      for (int nt = 0; nt < 4; ++nt) {
        float e = __expf(st[nt][r] - nm);
        st[nt][r] = e; rs += e;
      }
#pragma unroll
      for (int off = 1; off < 16; off <<= 1) rs += __shfl_xor(rs, off, 32);
      lrow[r] = lrow[r] * sc + rs;
#pragma unroll
      for (int v = 0; v < 8; ++v) acc[v][r] *= sc;
    }
#pragma unroll
    for (int nt = 0; nt < 4; ++nt)                         // C-layout -> LDS
#pragma unroll
      for (int r = 0; r < 8; ++r)
        sP[w][(r + 8 * hb) * 72 + nt * 16 + l15] = (__bf16)st[nt][r];
    __syncthreads();

    Frag fp[2];
#pragma unroll
    for (int kc = 0; kc < 2; ++kc) {                       // LDS -> A-layout
      const __bf16* p = &sP[w][l15 * 72 + kc * 32 + hb * 8];
      fp[kc].q[0] = *(const uint4*)p;
      fp[kc].q[1] = *(const uint4*)(p + 16);
    }
#pragma unroll
    for (int kc = 0; kc < 2; ++kc)
#pragma unroll
      for (int v = 0; v < 8; ++v) {
        Frag fv;
        const __bf16* p = &sVt[(v * 16 + l15) * 72 + kc * 32 + hb * 8];
        fv.q[0] = *(const uint4*)p; fv.q[1] = *(const uint4*)(p + 16);
        acc[v] = wmma_bf16(fp[kc], fv, acc[v]);
      }
    __syncthreads();
  }
#pragma unroll
  for (int v = 0; v < 8; ++v)
#pragma unroll
    for (int r = 0; r < 8; ++r) {
      int row = qb * 64 + w * 16 + r + 8 * hb;
      int col = head * 128 + v * 16 + l15;
      out[((size_t)b * S_LEN + row) * INTER_ + col] = acc[v][r] / lrow[r];
    }
}

// ================= merged = 0.5*(rms(attn)*ln1 + rms(y)*ln2), to bf16 =================
__global__ void merge_kernel(const float* __restrict__ attn, const float* __restrict__ y,
                             const float* __restrict__ ln1, const float* __restrict__ ln2,
                             __bf16* __restrict__ m) {
  __shared__ float r1[256], r2[256];
  int tok = blockIdx.x, t = threadIdx.x;
  const float* pa = attn + (size_t)tok * INTER_;
  const float* py = y + (size_t)tok * INTER_;
  float s1 = 0.f, s2 = 0.f;
#pragma unroll
  for (int ii = 0; ii < 8; ++ii) {
    float a = pa[t + ii * 256], b = py[t + ii * 256];
    s1 += a * a; s2 += b * b;
  }
  r1[t] = s1; r2[t] = s2; __syncthreads();
  for (int o = 128; o > 0; o >>= 1) {
    if (t < o) { r1[t] += r1[t + o]; r2[t] += r2[t + o]; }
    __syncthreads();
  }
  float f1 = rsqrtf(r1[0] / 2048.f + 1e-6f);
  float f2 = rsqrtf(r2[0] / 2048.f + 1e-6f);
#pragma unroll
  for (int ii = 0; ii < 8; ++ii) {
    int i = t + ii * 256;
    m[(size_t)tok * INTER_ + i] = (__bf16)(0.5f * (pa[i] * f1 * ln1[i] + py[i] * f2 * ln2[i]));
  }
}

// ================= host launch =================
extern "C" void kernel_launch(void* const* d_in, const int* in_sizes, int n_in,
                              void* d_out, int out_size, void* d_ws, size_t ws_size,
                              hipStream_t stream) {
  (void)in_sizes; (void)n_in; (void)out_size; (void)ws_size;
  const float* x      = (const float*)d_in[0];
  const float* W_in   = (const float*)d_in[1];
  const float* conv_w = (const float*)d_in[2];
  const float* conv_b = (const float*)d_in[3];
  const float* W_x    = (const float*)d_in[4];
  const float* dt_ln  = (const float*)d_in[5];
  const float* B_ln   = (const float*)d_in[6];
  const float* C_ln   = (const float*)d_in[7];
  const float* W_dt   = (const float*)d_in[8];
  const float* b_dt   = (const float*)d_in[9];
  const float* A_log  = (const float*)d_in[10];
  const float* D_par  = (const float*)d_in[11];
  const float* ln1    = (const float*)d_in[12];
  const float* ln2    = (const float*)d_in[13];
  const float* W_out  = (const float*)d_in[14];
  float* out = (float*)d_out;

  char* ws = (char*)d_ws;
  size_t off = 0;
  auto alloc = [&](size_t bytes) -> char* {
    char* p = ws + off; off += (bytes + 255) & ~(size_t)255; return p;
  };
  float*  proj   = (float*) alloc((size_t)TOK * LT_ * 4);
  __bf16* xbf    = (__bf16*)alloc((size_t)TOK * HID_ * 2);
  __bf16* winbf  = (__bf16*)alloc((size_t)LT_ * HID_ * 2);
  __bf16* woutbf = (__bf16*)alloc((size_t)HID_ * INTER_ * 2);
  __bf16* wxbf   = (__bf16*)alloc((size_t)128 * INTER_ * 2);
  float*  xc     = (float*) alloc((size_t)TOK * INTER_ * 4);
  __bf16* xcbf   = (__bf16*)alloc((size_t)TOK * INTER_ * 2);
  float*  ssmp   = (float*) alloc((size_t)TOK * 128 * 4);
  float*  dtn    = (float*) alloc((size_t)TOK * 64 * 4);
  float*  bn     = (float*) alloc((size_t)TOK * 16 * 4);
  float*  cn     = (float*) alloc((size_t)TOK * 16 * 4);
  float*  deltab = (float*) alloc((size_t)TOK * INTER_ * 4);
  float*  ybuf   = (float*) alloc((size_t)TOK * INTER_ * 4);
  float*  attnb  = (float*) alloc((size_t)TOK * INTER_ * 4);
  __bf16* qkv    = (__bf16*)alloc((size_t)TOK * 1792 * 2);
  __bf16* mbf    = (__bf16*)alloc((size_t)TOK * INTER_ * 2);

  { size_t n = (size_t)TOK * HID_;
    cvt_bf16<<<dim3((unsigned)((n + 255) / 256)), 256, 0, stream>>>(x, xbf, n); }
  { size_t n = (size_t)LT_ * HID_;
    cvt_bf16<<<dim3((unsigned)((n + 255) / 256)), 256, 0, stream>>>(W_in, winbf, n); }
  { size_t n = (size_t)HID_ * INTER_;
    cvt_bf16<<<dim3((unsigned)((n + 255) / 256)), 256, 0, stream>>>(W_out, woutbf, n); }
  pack_wx<<<dim3(128 * INTER_ / 256), 256, 0, stream>>>(W_x, wxbf);

  gemm_bf16_tn<<<dim3(LT_ / 128, TOK / 128), 256, 0, stream>>>(xbf, winbf, proj, TOK, LT_, HID_);

  pack_qkv<<<dim3((unsigned)((size_t)TOK * 1792 / 256)), 256, 0, stream>>>(proj, qkv);
  conv_silu<<<dim3(INTER_ / 256, S_LEN, BATCH), 256, 0, stream>>>(proj, conv_w, conv_b, xc);
  { size_t n = (size_t)TOK * INTER_;
    cvt_bf16<<<dim3((unsigned)((n + 255) / 256)), 256, 0, stream>>>(xc, xcbf, n); }

  gemm_bf16_tn<<<dim3(1, TOK / 128), 256, 0, stream>>>(xcbf, wxbf, ssmp, TOK, 128, INTER_);
  dtbc_norm<<<dim3(TOK), 128, 0, stream>>>(ssmp, dt_ln, B_ln, C_ln, dtn, bn, cn);
  delta_kernel<<<dim3(TOK), 256, 0, stream>>>(dtn, W_dt, b_dt, deltab);
  scan_kernel<<<dim3(INTER_ / 256, BATCH), 256, 0, stream>>>(deltab, bn, cn, xc, proj,
                                                             A_log, D_par, ybuf);
  attn_kernel<<<dim3(S_LEN / 64, NH, BATCH), 128, 0, stream>>>(qkv, attnb);
  merge_kernel<<<dim3(TOK), 256, 0, stream>>>(attnb, ybuf, ln1, ln2, mbf);

  gemm_bf16_tn<<<dim3(HID_ / 128, TOK / 128), 256, 0, stream>>>(mbf, woutbf, out, TOK, HID_, INTER_);
}
